// GAE_2422361555220
// MI455X (gfx1250) — compile-verified, matrix operand order checked by reading, not run on hardware
//
#include <hip/hip_runtime.h>
#include <hip/hip_bf16.h>

typedef __bf16 bf16;
typedef __bf16 v16bf __attribute__((ext_vector_type(16)));
typedef float  v8f   __attribute__((ext_vector_type(8)));
typedef unsigned int u32x4 __attribute__((ext_vector_type(4)));
typedef int          i32x4 __attribute__((ext_vector_type(4)));
typedef int          i32x8 __attribute__((ext_vector_type(8)));

#if __has_builtin(__builtin_amdgcn_tensor_load_to_lds) && __has_builtin(__builtin_amdgcn_s_wait_tensorcnt)
#define GFN_USE_TDM 1
#else
#define GFN_USE_TDM 0
#endif

static inline int cdiv(long a, long b){ return (int)((a + b - 1) / b); }

// ---------------- elementwise / graph helper kernels ----------------

__global__ void k_zero(float* __restrict__ p, long n){
  long i = (long)blockIdx.x * blockDim.x + threadIdx.x;
  if (i < n) p[i] = 0.f;
}

__global__ void k_f2b(const float* __restrict__ x, bf16* __restrict__ y, long n){
  long i = (long)blockIdx.x * blockDim.x + threadIdx.x;
  if (i < n) y[i] = (bf16)x[i];
}

__global__ void k_add3b(const float* __restrict__ a, const float* __restrict__ b,
                        const float* __restrict__ c, bf16* __restrict__ y, long n){
  long i = (long)blockIdx.x * blockDim.x + threadIdx.x;
  if (i < n) y[i] = (bf16)(a[i] + b[i] + c[i]);
}

__global__ void k_edge_deg(const int* __restrict__ src, const int* __restrict__ dst,
                           float* __restrict__ degO, float* __restrict__ degI, int E){
  int e = blockIdx.x * blockDim.x + threadIdx.x;
  if (e < E){
    atomicAdd(&degO[src[e]], 1.f);
    atomicAdd(&degI[dst[e]], 1.f);
  }
}

__global__ void k_scale_rows_b(const float* __restrict__ x, const float* __restrict__ deg,
                               bf16* __restrict__ y, int C){
  int i = blockIdx.y;
  int j = blockIdx.x * blockDim.x + threadIdx.x;
  if (j < C){
    long o = (long)i * C + j;
    y[o] = (bf16)(x[o] * rsqrtf(fmaxf(deg[i], 1.f)));
  }
}

__global__ void k_edge_agg(const int* __restrict__ src, const int* __restrict__ dst,
                           const float* __restrict__ h, float* __restrict__ agg, int C){
  int e = blockIdx.x;
  int j = threadIdx.x;
  atomicAdd(&agg[(long)dst[e] * C + j], h[(long)src[e] * C + j]);
}

__global__ void k_finish(const float* __restrict__ agg, const float* __restrict__ degI,
                         const float* __restrict__ bias, float* __restrict__ y,
                         int C, int relu){
  int i = blockIdx.y;
  int j = blockIdx.x * blockDim.x + threadIdx.x;
  if (j < C){
    long o = (long)i * C + j;
    float v = agg[o] * rsqrtf(fmaxf(degI[i], 1.f)) + bias[j];
    if (relu) v = fmaxf(v, 0.f);
    y[o] = v;
  }
}

__global__ void k_softmax64(float* __restrict__ z){
  int row = blockIdx.x, l = threadIdx.x;
  long o = (long)row * 64;
  float a = z[o + l], b = z[o + 32 + l];
  float m = fmaxf(a, b);
  #pragma unroll
  for (int s = 16; s > 0; s >>= 1) m = fmaxf(m, __shfl_xor(m, s, 32));
  float ea = __expf(a - m), eb = __expf(b - m);
  float t = ea + eb;
  #pragma unroll
  for (int s = 16; s > 0; s >>= 1) t += __shfl_xor(t, s, 32);
  float inv = 1.f / t;
  z[o + l] = ea * inv;
  z[o + 32 + l] = eb * inv;
}

__global__ void k_build_A(const float* __restrict__ adj_r, bf16* __restrict__ A, int n){
  int i = blockIdx.y;
  int j = blockIdx.x * blockDim.x + threadIdx.x;
  if (j < n){
    bool p = adj_r[(long)i * n + j] > 0.4f;
    bool q = adj_r[(long)j * n + i] > 0.4f;
    float v = ((p || q) ? 1.f : 0.f) + ((i == j) ? 1.f : 0.f);
    A[(long)i * n + j] = (bf16)v;
  }
}

__global__ void k_rowsum_b(const bf16* __restrict__ A, float* __restrict__ deg, int n){
  __shared__ float red[256];
  int i = blockIdx.x;
  float s = 0.f;
  for (int j = threadIdx.x; j < n; j += 256) s += (float)A[(long)i * n + j];
  red[threadIdx.x] = s;
  __syncthreads();
  #pragma unroll
  for (int o = 128; o > 0; o >>= 1){
    if (threadIdx.x < (unsigned)o) red[threadIdx.x] += red[threadIdx.x + o];
    __syncthreads();
  }
  if (threadIdx.x == 0) deg[i] = red[0];
}

// ---------------- TDM tile fetch (CDNA5 Tensor Data Mover) ----------------
// 2D tile: tile_dim0 = 32 bf16 (64B rows), tile_dim1 = rows, row stride = stride elems.
// LDS layout padded: 16B pad after every 64B row -> pad_interval enc 3 (16 DWORDs),
// pad_amount enc 3 (4 DWORDs). data_size enc 1 (2 bytes). D# per ISA 08_async_tensor §8.
#if GFN_USE_TDM
__device__ __forceinline__ void tdm_tile(const bf16* g, unsigned lds_byte,
                                         int rows, long stride_elems)
{
  unsigned long long ga = (unsigned long long)(uintptr_t)g;
  u32x4 g0;
  g0[0] = 1u;                                            // count=1
  g0[1] = lds_byte;                                      // lds_addr
  g0[2] = (unsigned)(ga & 0xffffffffu);                  // global_addr[31:0]
  g0[3] = (unsigned)((ga >> 32) & 0x1ffffffu) | (2u<<30);// addr[56:32] | type=2
  i32x8 g1;
  g1[0] = (int)((1u << 16) | (1u << 20) | (3u << 22) | (3u << 25));
  g1[1] = (int)(32u << 16);                              // tensor_dim0 = 32
  g1[2] = (int)(((unsigned)rows & 0xffffu) << 16);       // tensor_dim1 = rows
  g1[3] = (int)(32u << 16);                              // tile_dim0 = 32
  g1[4] = (int)((unsigned)rows & 0xffffu);               // tile_dim1 = rows (tile_dim2=0)
  unsigned long long st = (unsigned long long)stride_elems;
  g1[5] = (int)(st & 0xffffffffu);                       // dim0 stride lo
  g1[6] = (int)((st >> 32) & 0xffffu);                   // dim0 stride hi
  g1[7] = 0;
  i32x4 gz4; gz4[0]=0; gz4[1]=0; gz4[2]=0; gz4[3]=0;
  i32x8 gz8; gz8[0]=0; gz8[1]=0; gz8[2]=0; gz8[3]=0; gz8[4]=0; gz8[5]=0; gz8[6]=0; gz8[7]=0;
  // 6-arg form on this toolchain (clang-23 / therock headers)
  __builtin_amdgcn_tensor_load_to_lds(g0, g1, gz4, gz4, gz8, 0);
}
#endif

// ---------------- WMMA bf16 GEMM ----------------
__device__ __forceinline__ void wmma_step(const bf16* Asb, const bf16* Bsb,
                                          int wm, int wn, int mloc, int half,
                                          v8f (&c)[2][2])
{
  constexpr int LPAD = 40;
  v16bf af[2], bfr[2];
  #pragma unroll
  for (int im = 0; im < 2; ++im) {
    const bf16* ap = Asb + (wm * 32 + im * 16 + mloc) * LPAD + half * 8;
    ((uint4*)&af[im])[0] = *(const uint4*)(ap);
    ((uint4*)&af[im])[1] = *(const uint4*)(ap + 16);
  }
  #pragma unroll
  for (int in = 0; in < 2; ++in) {
    const bf16* bp = Bsb + (wn * 32 + in * 16 + mloc) * LPAD + half * 8;
    ((uint4*)&bfr[in])[0] = *(const uint4*)(bp);
    ((uint4*)&bfr[in])[1] = *(const uint4*)(bp + 16);
  }
  #pragma unroll
  for (int im = 0; im < 2; ++im)
    #pragma unroll
    for (int in = 0; in < 2; ++in)
      c[im][in] = __builtin_amdgcn_wmma_f32_16x16x32_bf16(
          false, af[im], false, bfr[in], (short)0, c[im][in], false, false);
}

// C[M,Nn] = A[M,K] * op(B) (+bias) (+existing C) (relu), fp32 accumulate, bf16 inputs.
// TB=false: B row-major [K,Nn].  TB=true: B row-major [Nn,K] (compute A*B^T).
template<int WM, int WN, bool TB>
__global__ __launch_bounds__(WM*WN*32)
void k_gemm(const bf16* __restrict__ A, const bf16* __restrict__ B,
            const float* __restrict__ bias,
            float* __restrict__ Cf, bf16* __restrict__ Cb,
            int M, int Nn, int K, int relu, int accum)
{
  constexpr int BM = 32 * WM, BN = 32 * WN, KT = 32, LPAD = 40, T = WM * WN * 32;
  constexpr int NBUF = (GFN_USE_TDM && TB) ? 2 : 1;
  __shared__ __align__(16) bf16 As[NBUF * BM * LPAD];
  __shared__ __align__(16) bf16 Bs[NBUF * BN * LPAD];

  const int tid  = threadIdx.x;
  const int lane = tid & 31, wid = tid >> 5;
  const int wm = wid / WN, wn = wid % WN;
  const long m0 = (long)blockIdx.x * BM;
  const long n0 = (long)blockIdx.y * BN;
  const int mloc = lane & 15, half = lane >> 4;

  v8f c[2][2];
  #pragma unroll
  for (int im = 0; im < 2; ++im)
    #pragma unroll
    for (int in = 0; in < 2; ++in)
      #pragma unroll
      for (int r = 0; r < 8; ++r) c[im][in][r] = 0.f;

#if GFN_USE_TDM
  const unsigned lds_as = (unsigned)(uintptr_t)(void*)&As[0];
  const unsigned lds_bs = (unsigned)(uintptr_t)(void*)&Bs[0];
  if (TB) {
    // ---- double-buffered, both tiles via Tensor Data Mover ----
    if (wid == 0) {
      tdm_tile(A + m0 * K, lds_as, BM, K);
      tdm_tile(B + n0 * K, lds_bs, BN, K);
    }
    int buf = 0;
    for (long kk = 0; kk < K; kk += KT) {
      const bool more = (kk + KT) < K;
      if (wid == 0) {
        if (more) {
          tdm_tile(A + m0 * K + kk + KT, lds_as + (unsigned)((buf ^ 1) * BM * LPAD * 2), BM, K);
          tdm_tile(B + n0 * K + kk + KT, lds_bs + (unsigned)((buf ^ 1) * BN * LPAD * 2), BN, K);
          __builtin_amdgcn_s_wait_tensorcnt(2);   // current buffer's 2 DMAs done
        } else {
          __builtin_amdgcn_s_wait_tensorcnt(0);
        }
      }
      __syncthreads();
      wmma_step(&As[buf * BM * LPAD], &Bs[buf * BN * LPAD], wm, wn, mloc, half, c);
      __syncthreads();          // everyone done reading before buf is re-filled
      buf ^= 1;
    }
  } else {
    // ---- A tile via TDM, B tile VALU-transposed (small GEMMs) ----
    for (long kk = 0; kk < K; kk += KT) {
      if (wid == 0) tdm_tile(A + m0 * K + kk, lds_as, BM, K);
      for (int idx = tid; idx < KT * BN; idx += T) {
        int k = idx / BN, n = idx % BN;           // coalesced along n
        Bs[n * LPAD + k] = B[(kk + k) * (long)Nn + n0 + n];
      }
      if (wid == 0) __builtin_amdgcn_s_wait_tensorcnt(0);
      __syncthreads();
      wmma_step(As, Bs, wm, wn, mloc, half, c);
      __syncthreads();
    }
  }
#else
  // ---- fallback: synchronous VALU staging ----
  for (long kk = 0; kk < K; kk += KT) {
    #pragma unroll 2
    for (int idx = tid; idx < BM * 4; idx += T) {
      int r = idx >> 2, q = idx & 3;
      uint4 v = *(const uint4*)(A + (m0 + r) * K + kk + q * 8);
      *(uint4*)(&As[r * LPAD + q * 8]) = v;
    }
    if (TB) {
      #pragma unroll 2
      for (int idx = tid; idx < BN * 4; idx += T) {
        int r = idx >> 2, q = idx & 3;
        uint4 v = *(const uint4*)(B + (n0 + r) * K + kk + q * 8);
        *(uint4*)(&Bs[r * LPAD + q * 8]) = v;
      }
    } else {
      for (int idx = tid; idx < KT * BN; idx += T) {
        int k = idx / BN, n = idx % BN;
        Bs[n * LPAD + k] = B[(kk + k) * (long)Nn + n0 + n];
      }
    }
    if (kk + KT < K)
      __builtin_prefetch(A + (m0 + (tid & (BM - 1))) * K + kk + KT, 0, 0);
    __syncthreads();
    wmma_step(As, Bs, wm, wn, mloc, half, c);
    __syncthreads();
  }
#endif

  // epilogue: C/D layout -> row = r + 8*half, col = lane&15
  #pragma unroll
  for (int im = 0; im < 2; ++im)
    #pragma unroll
    for (int in = 0; in < 2; ++in)
      #pragma unroll
      for (int r = 0; r < 8; ++r) {
        long row = m0 + wm * 32 + im * 16 + half * 8 + r;
        long col = n0 + wn * 32 + in * 16 + mloc;
        float v = c[im][in][r];
        if (accum) v += Cf[row * Nn + col];
        if (bias)  v += bias[col];
        if (relu)  v = fmaxf(v, 0.f);
        if (Cf) Cf[row * Nn + col] = v;
        if (Cb) Cb[row * Nn + col] = (bf16)v;
      }
}

static void gemm(hipStream_t s, const bf16* A, const bf16* B, const float* bias,
                 float* Cf, bf16* Cb, int M, int Nn, int K,
                 bool transB, bool relu, bool accum)
{
  int rl = relu ? 1 : 0, ac = accum ? 1 : 0;
  if (Nn % 128 == 0) {
    dim3 g(M / 64, Nn / 128);
    if (transB) k_gemm<2,4,true ><<<g, 256, 0, s>>>(A,B,bias,Cf,Cb,M,Nn,K,rl,ac);
    else        k_gemm<2,4,false><<<g, 256, 0, s>>>(A,B,bias,Cf,Cb,M,Nn,K,rl,ac);
  } else if (Nn % 64 == 0) {
    dim3 g(M / 64, Nn / 64);
    if (transB) k_gemm<2,2,true ><<<g, 128, 0, s>>>(A,B,bias,Cf,Cb,M,Nn,K,rl,ac);
    else        k_gemm<2,2,false><<<g, 128, 0, s>>>(A,B,bias,Cf,Cb,M,Nn,K,rl,ac);
  } else { // Nn == 32
    dim3 g(M / 128, Nn / 32);
    if (transB) k_gemm<4,1,true ><<<g, 128, 0, s>>>(A,B,bias,Cf,Cb,M,Nn,K,rl,ac);
    else        k_gemm<4,1,false><<<g, 128, 0, s>>>(A,B,bias,Cf,Cb,M,Nn,K,rl,ac);
  }
}

// ---------------- orchestration ----------------

extern "C" void kernel_launch(void* const* d_in, const int* in_sizes, int n_in,
                              void* d_out, int out_size, void* d_ws, size_t ws_size,
                              hipStream_t stream)
{
  (void)in_sizes; (void)n_in; (void)out_size; (void)ws_size;
  const int Nn = 4096, E = 65536, Hh = 2048;
  const int Dv[3] = {512, 256, 128};

  const float* data[3] = {(const float*)d_in[0], (const float*)d_in[1], (const float*)d_in[2]};
  const int*   src[3]  = {(const int*)d_in[3], (const int*)d_in[6], (const int*)d_in[9]};
  const int*   dst[3]  = {(const int*)d_in[4], (const int*)d_in[7], (const int*)d_in[10]};
  const float* adj[3]  = {(const float*)d_in[5], (const float*)d_in[8], (const float*)d_in[11]};
  const float* wv[3][2] = {
    {(const float*)d_in[12], (const float*)d_in[14]},
    {(const float*)d_in[16], (const float*)d_in[18]},
    {(const float*)d_in[20], (const float*)d_in[22]}};
  const float* bv[3][2] = {
    {(const float*)d_in[13], (const float*)d_in[15]},
    {(const float*)d_in[17], (const float*)d_in[19]},
    {(const float*)d_in[21], (const float*)d_in[23]}};
  const float* wm[2] = {(const float*)d_in[24], (const float*)d_in[26]};
  const float* bm[2] = {(const float*)d_in[25], (const float*)d_in[27]};
  const float* fw[3] = {(const float*)d_in[28], (const float*)d_in[29], (const float*)d_in[30]};
  const float* gw1 = (const float*)d_in[31];
  const float* gb1 = (const float*)d_in[32];
  const float* gw2 = (const float*)d_in[33];
  const float* gb2 = (const float*)d_in[34];

  float* adj_r   = (float*)d_out;                 // [N,N]
  float* adj_rec = adj_r + (size_t)Nn * Nn;       // [N,N]
  float* feat_o  = adj_rec + (size_t)Nn * Nn;     // [N,32]

  char* wp = (char*)d_ws;
  auto carve = [&](size_t bytes) -> void* {
    void* r = (void*)wp; wp += (bytes + 255) & ~(size_t)255; return r;
  };
  bf16*  big1  = (bf16*)carve((size_t)Nn * Nn * 2);    // adjin, later consensus A
  bf16*  w1b   = (bf16*)carve((size_t)Hh * Nn * 2);
  bf16*  w2b   = (bf16*)carve((size_t)Nn * Hh * 2);
  bf16*  T1b   = (bf16*)carve((size_t)Nn * Hh * 2);
  bf16*  Xsb   = (bf16*)carve((size_t)Nn * 512 * 2);
  bf16*  Wb    = (bf16*)carve((size_t)512 * 128 * 2);
  float* hbuf  = (float*)carve((size_t)Nn * 128 * 4);
  float* aggb  = (float*)carve((size_t)Nn * 128 * 4);
  float* ybuf  = (float*)carve((size_t)Nn * 128 * 4);
  bf16*  fb[3];
  for (int v = 0; v < 3; ++v) fb[v] = (bf16*)carve((size_t)Nn * 64 * 2);
  float* zb    = (float*)carve((size_t)Nn * 64 * 4);
  bf16*  zsb   = (bf16*)carve((size_t)Nn * 64 * 2);
  bf16*  hsb   = (bf16*)carve((size_t)Nn * 64 * 2);
  float* f1b   = (float*)carve((size_t)Nn * 64 * 4);
  bf16*  featb = (bf16*)carve((size_t)Nn * 32 * 2);
  float* degO  = (float*)carve((size_t)Nn * 4);
  float* degI  = (float*)carve((size_t)Nn * 4);

  // ---- per-view GraphConv stacks ----
  for (int v = 0; v < 3; ++v) {
    int D = Dv[v];
    k_zero<<<cdiv(Nn,256),256,0,stream>>>(degO, Nn);
    k_zero<<<cdiv(Nn,256),256,0,stream>>>(degI, Nn);
    k_edge_deg<<<cdiv(E,256),256,0,stream>>>(src[v], dst[v], degO, degI, E);

    k_scale_rows_b<<<dim3(cdiv(D,128),Nn),128,0,stream>>>(data[v], degO, Xsb, D);
    k_f2b<<<cdiv((long)D*128,256),256,0,stream>>>(wv[v][0], Wb, (long)D*128);
    gemm(stream, Xsb, Wb, nullptr, hbuf, nullptr, Nn, 128, D, false, false, false);
    k_zero<<<cdiv((long)Nn*128,256),256,0,stream>>>(aggb, (long)Nn*128);
    k_edge_agg<<<E,128,0,stream>>>(src[v], dst[v], hbuf, aggb, 128);
    k_finish<<<dim3(1,Nn),128,0,stream>>>(aggb, degI, bv[v][0], ybuf, 128, 1);

    k_scale_rows_b<<<dim3(1,Nn),128,0,stream>>>(ybuf, degO, Xsb, 128);
    k_f2b<<<cdiv(128*64,256),256,0,stream>>>(wv[v][1], Wb, 128*64);
    gemm(stream, Xsb, Wb, nullptr, hbuf, nullptr, Nn, 64, 128, false, false, false);
    k_zero<<<cdiv((long)Nn*64,256),256,0,stream>>>(aggb, (long)Nn*64);
    k_edge_agg<<<E,64,0,stream>>>(src[v], dst[v], hbuf, aggb, 64);
    k_finish<<<dim3(1,Nn),128,0,stream>>>(aggb, degI, bv[v][1], ybuf, 64, 0);
    k_f2b<<<cdiv((long)Nn*64,256),256,0,stream>>>(ybuf, fb[v], (long)Nn*64);
  }

  // ---- fusion: z = softmax(f1@fw1 + f2@fw2 + f3@fw3) ----
  for (int v = 0; v < 3; ++v) {
    k_f2b<<<cdiv(64*64,256),256,0,stream>>>(fw[v], Wb, 64*64);
    gemm(stream, fb[v], Wb, nullptr, zb, nullptr, Nn, 64, 64, false, false, v > 0);
  }
  k_softmax64<<<Nn,32,0,stream>>>(zb);

  // ---- GFN (dominant WMMA GEMMs, TDM-fed) ----
  k_add3b<<<cdiv((long)Nn*Nn,256),256,0,stream>>>(adj[0], adj[1], adj[2], big1, (long)Nn*Nn);
  k_f2b<<<cdiv((long)Hh*Nn,256),256,0,stream>>>(gw1, w1b, (long)Hh*Nn);
  k_f2b<<<cdiv((long)Nn*Hh,256),256,0,stream>>>(gw2, w2b, (long)Nn*Hh);
  gemm(stream, big1, w1b, gb1, nullptr, T1b, Nn, Hh, Nn, true, true,  false);
  gemm(stream, T1b,  w2b, gb2, adj_r,   nullptr, Nn, Nn, Hh, true, false, false);

  // ---- consensus graph (A symmetric) ----
  k_build_A<<<dim3(cdiv(Nn,256),Nn),256,0,stream>>>(adj_r, big1, Nn);
  k_rowsum_b<<<Nn,256,0,stream>>>(big1, degO, Nn);

  k_scale_rows_b<<<dim3(1,Nn),128,0,stream>>>(zb, degO, zsb, 64);
  k_f2b<<<cdiv(64*64,256),256,0,stream>>>(wm[0], Wb, 64*64);
  gemm(stream, zsb,  Wb,  nullptr, nullptr, hsb, Nn, 64, 64, false, false, false);
  gemm(stream, big1, hsb, nullptr, aggb, nullptr, Nn, 64, Nn, false, false, false);
  k_finish<<<dim3(1,Nn),128,0,stream>>>(aggb, degO, bm[0], f1b, 64, 1);

  k_scale_rows_b<<<dim3(1,Nn),128,0,stream>>>(f1b, degO, zsb, 64);
  k_f2b<<<cdiv(64*32,256),256,0,stream>>>(wm[1], Wb, 64*32);
  gemm(stream, zsb,  Wb,  nullptr, nullptr, hsb, Nn, 32, 64, false, false, false);
  gemm(stream, big1, hsb, nullptr, aggb, nullptr, Nn, 32, Nn, false, false, false);
  k_finish<<<dim3(1,Nn),128,0,stream>>>(aggb, degO, bm[1], feat_o, 32, 0);
  k_f2b<<<cdiv((long)Nn*32,256),256,0,stream>>>(feat_o, featb, (long)Nn*32);

  // ---- adj_rec = feat @ feat^T ----
  gemm(stream, featb, featb, nullptr, adj_rec, nullptr, Nn, Nn, 32, true, false, false);
}
